// VectorQuantizer_19859928777081
// MI455X (gfx1250) — compile-verified, matrix operand order checked by reading, not run on hardware
//
#include <hip/hip_runtime.h>
#include <hip/hip_fp16.h>

typedef __attribute__((ext_vector_type(16))) _Float16 v16h;
typedef __attribute__((ext_vector_type(8)))  _Float16 v8h;
typedef __attribute__((ext_vector_type(8)))  float    v8f;
typedef __attribute__((ext_vector_type(4)))  unsigned int u32x4;
typedef __attribute__((ext_vector_type(8)))  int          i32x8;
typedef __attribute__((ext_vector_type(4)))  int          i32x4;

#define B_    8
#define N_    8192
#define DIM_  256
#define K_    4096
#define CDIM_ 1024
#define ROWS_ (B_*N_)   // 65536

// ---- WMMA fragment load: lane-local 16 halves from a row-major row pointer ----
// 16-bit A 16x32 layout: lanes 0-15 take K=[kb..kb+7],[kb+16..kb+23] with kb=k0;
// lanes 16-31 use kb=k0+8. Caller passes kb = k0 + ((lane&16)?8:0).
static __device__ __forceinline__ v16h load_frag(const _Float16* rowp, int kb) {
  v8h lo = *(const v8h*)(rowp + kb);
  v8h hi = *(const v8h*)(rowp + kb + 16);
  v16h r;
#pragma unroll
  for (int i = 0; i < 8; ++i) { r[i] = lo[i]; r[8 + i] = hi[i]; }
  return r;
}

#define WMMA(a, b, c) __builtin_amdgcn_wmma_f32_16x16x32_f16(false, (a), false, (b), (short)0, (c), false, false)

// ---- TDM: 2D f16 tile (rows x dim0, row stride in elements) global -> LDS ----
static __device__ __forceinline__ void tdm_load_2d(unsigned int lds_off, const void* gptr,
                                                   unsigned int dim0, unsigned int rows,
                                                   unsigned int stride, unsigned int trows) {
  unsigned long long ga = (unsigned long long)(uintptr_t)gptr;
  u32x4 g0;
  g0[0] = 1u;                                              // count=1 (valid), user mode
  g0[1] = lds_off;                                         // LDS byte address
  g0[2] = (unsigned int)ga;                                // global_addr[31:0]
  g0[3] = (unsigned int)((ga >> 32) & 0x1FFFFFFu) | (2u << 30); // addr[56:32] | type=2
  i32x8 g1;
  g1[0] = 0x00010000;                                      // data_size=1 (2 bytes)
  g1[1] = (int)((dim0 & 0xffffu) << 16);                   // tensor_dim0 lo16
  g1[2] = (int)(((dim0 >> 16) & 0xffffu) | ((trows & 0xffffu) << 16)); // dim0 hi | dim1 lo
  g1[3] = (int)(((trows >> 16) & 0xffffu) | ((dim0 & 0xffffu) << 16)); // dim1 hi | tile_dim0
  g1[4] = (int)(rows & 0xffffu);                           // tile_dim1 (tile_dim2=0)
  g1[5] = (int)stride;                                     // tensor_dim0_stride lo32
  g1[6] = 0; g1[7] = 0;
  i32x4 g2 = {0, 0, 0, 0}, g3 = {0, 0, 0, 0};
  i32x8 gx = {0, 0, 0, 0, 0, 0, 0, 0};
  __builtin_amdgcn_tensor_load_to_lds(g0, g1, g2, g3, gx, 0);
}

// ---- TDM gather mode: 16 rows of (dim0 f16) selected by 16-bit indices ----
static __device__ __forceinline__ void tdm_gather16(unsigned int lds_off, const void* gptr,
                                                    unsigned int dim0, unsigned int trows,
                                                    unsigned int stride, i32x4 g2, i32x4 g3) {
  unsigned long long ga = (unsigned long long)(uintptr_t)gptr;
  u32x4 g0;
  g0[0] = 1u | (1u << 31);                                 // count=1, gather_mode, 16b idx
  g0[1] = lds_off;
  g0[2] = (unsigned int)ga;
  g0[3] = (unsigned int)((ga >> 32) & 0x1FFFFFFu) | (2u << 30);
  i32x8 g1;
  g1[0] = 0x00010000;                                      // data_size=2B
  g1[1] = (int)((dim0 & 0xffffu) << 16);
  g1[2] = (int)(((dim0 >> 16) & 0xffffu) | ((trows & 0xffffu) << 16));
  g1[3] = (int)(((trows >> 16) & 0xffffu) | ((dim0 & 0xffffu) << 16));
  g1[4] = 16;                                              // tile_dim1 = #valid indices
  g1[5] = (int)stride;
  g1[6] = 0; g1[7] = 0;
  i32x8 gx = {0, 0, 0, 0, 0, 0, 0, 0};
  __builtin_amdgcn_tensor_load_to_lds(g0, g1, g2, g3, gx, 0);
}

// ---------------- prep: embed -> f16 + per-code squared norm ----------------
__global__ void k_prep_embed(const float* __restrict__ e, _Float16* __restrict__ eh,
                             float* __restrict__ esq) {
  __shared__ float s[256];
  int k = blockIdx.x, t = threadIdx.x;
  float acc = 0.f;
#pragma unroll
  for (int i = 0; i < 4; ++i) {
    int c = t + i * 256;
    float v = e[(long)k * CDIM_ + c];
    eh[(long)k * CDIM_ + c] = (_Float16)v;
    acc += v * v;
  }
  s[t] = acc; __syncthreads();
  for (int o = 128; o > 0; o >>= 1) { if (t < o) s[t] += s[t + o]; __syncthreads(); }
  if (t == 0) esq[k] = s[0];
}

// ---------------- prep: transpose+convert weights to f16 ----------------
__global__ void k_prep_w(const float* __restrict__ wpre, const float* __restrict__ wpost,
                         _Float16* __restrict__ wpreT, _Float16* __restrict__ wpostT) {
  int tid = blockIdx.x * 256 + threadIdx.x;   // 0..262143
  { int r = tid / CDIM_, c = tid % CDIM_;     // w_pre: (DIM x CDIM)
    wpreT[(long)c * DIM_ + r] = (_Float16)wpre[tid]; }
  { int r = tid / DIM_,  c = tid % DIM_;      // w_post: (CDIM x DIM)
    wpostT[(long)c * CDIM_ + r] = (_Float16)wpost[tid]; }
}

// ---------------- LayerNorm + exact GELU -> f16 ----------------
__global__ void k_ln_gelu(const float* __restrict__ x, const float* __restrict__ g,
                          const float* __restrict__ b, _Float16* __restrict__ h) {
  __shared__ float s1[256], s2[256];
  int row = blockIdx.x, t = threadIdx.x;
  float v = x[(long)row * DIM_ + t];
  s1[t] = v; s2[t] = v * v;
  __syncthreads();
  for (int o = 128; o > 0; o >>= 1) {
    if (t < o) { s1[t] += s1[t + o]; s2[t] += s2[t + o]; }
    __syncthreads();
  }
  float mu  = s1[0] * (1.0f / DIM_);
  float var = s2[0] * (1.0f / DIM_) - mu * mu;
  float xn  = (v - mu) * rsqrtf(var + 1e-5f) * g[t] + b[t];
  float ge  = 0.5f * xn * (1.0f + erff(xn * 0.70710678118654752f));
  h[(long)row * DIM_ + t] = (_Float16)ge;
}

// ---------------- GEMM1: z_e = h(65536x256) @ w_pre(256x1024) + b ----------------
__global__ void k_gemm_pre(const _Float16* __restrict__ h, const _Float16* __restrict__ wpreT,
                           const float* __restrict__ bpre, _Float16* __restrict__ ze) {
  __shared__ _Float16 As[16 * DIM_];          // 8 KB A tile
  int t = threadIdx.x;                        // 256 threads = 8 waves
  int wave = t >> 5, lane = t & 31;
  long rowbase = (long)blockIdx.x * 16;
  if (t < 32) {                               // wave 0 issues the TDM stage
    tdm_load_2d((unsigned int)(uintptr_t)As, h + rowbase * DIM_, DIM_, 16, DIM_, 16);
    __builtin_amdgcn_s_wait_tensorcnt(0);
  }
  __syncthreads();
  int r0 = lane & 15;
  int m0 = (lane & 16) ? 8 : 0;
  int kboff = m0;
  const _Float16* arow = As + r0 * DIM_;
#pragma unroll 1
  for (int gg = 0; gg < 2; ++gg) {            // 8 N-tiles per wave = 2 groups of 4
    asm volatile("" ::: "memory");            // keep A fragments in LDS, not regs/scratch
    int col0 = (wave * 8 + gg * 4) * 16 + r0;
    const _Float16* brow = wpreT + (long)col0 * DIM_;
    v8f cc[4] = {};
#pragma unroll 2
    for (int k0 = 0; k0 < DIM_; k0 += 32) {
      v16h a = load_frag(arow, k0 + kboff);
#pragma unroll
      for (int u = 0; u < 4; ++u) {
        v16h b = load_frag(brow + u * 16 * DIM_, k0 + kboff);
        cc[u] = WMMA(a, b, cc[u]);
      }
    }
#pragma unroll
    for (int u = 0; u < 4; ++u) {
      int col = col0 + u * 16;
      float bias = bpre[col];
#pragma unroll
      for (int r = 0; r < 8; ++r)
        ze[(rowbase + r + m0) * CDIM_ + col] = (_Float16)(cc[u][r] + bias);
    }
  }
}

// ---------------- distance GEMM + running argmin ----------------
// argmin_k dist = argmin_k ( e_sq[k] - 2 * z.e_k )  (z_sq is row-constant)
__global__ void k_dist_argmin(const _Float16* __restrict__ ze, const _Float16* __restrict__ eh,
                              const float* __restrict__ esq, int* __restrict__ out_idx,
                              int* __restrict__ ws_idx) {
  __shared__ _Float16 As[16 * CDIM_];         // 32 KB A tile (16 rows of z_e)
  __shared__ float sval[4 * 16];
  __shared__ int   sidx[4 * 16];
  int t = threadIdx.x;                        // 128 threads = 4 waves
  int wave = t >> 5, lane = t & 31;
  long rowbase = (long)blockIdx.x * 16;
  if (t < 32) {                               // TDM-stage the z_e tile
    tdm_load_2d((unsigned int)(uintptr_t)As, ze + rowbase * CDIM_, CDIM_, 16, CDIM_, 16);
    __builtin_amdgcn_s_wait_tensorcnt(0);
  }
  __syncthreads();
  int r0 = lane & 15;
  int m0 = (lane & 16) ? 8 : 0;
  int kboff = m0;
  const _Float16* arow = As + r0 * CDIM_;
  float best[8]; int bidx[8];
#pragma unroll
  for (int r = 0; r < 8; ++r) { best[r] = 3.4e38f; bidx[r] = 0; }
#pragma unroll 1
  for (int g = 0; g < 16; ++g) {              // 16 groups x 4 N-tiles = 64 tiles/wave
    asm volatile("" ::: "memory");            // prevent hoisting A frags across groups
    int code0 = (g * 16 + wave * 4) * 16 + r0;
    const _Float16* brow = eh + (long)code0 * CDIM_;
    if (g < 15) {                             // warm next group's B rows into cache
      const _Float16* bnext = brow + (long)256 * CDIM_;
#pragma unroll
      for (int u = 0; u < 4; ++u) __builtin_prefetch(bnext + u * 16 * CDIM_, 0, 1);
    }
    v8f cc[4] = {};
#pragma unroll 2
    for (int k0 = 0; k0 < CDIM_; k0 += 32) {
      v16h a = load_frag(arow, k0 + kboff);   // one A fragment feeds 4 WMMAs
#pragma unroll
      for (int u = 0; u < 4; ++u) {
        v16h b = load_frag(brow + u * 16 * CDIM_, k0 + kboff);
        cc[u] = WMMA(a, b, cc[u]);
      }
    }
#pragma unroll
    for (int u = 0; u < 4; ++u) {
      int code = code0 + u * 16;
      float eq = esq[code];
#pragma unroll
      for (int r = 0; r < 8; ++r) {
        float d = eq - 2.0f * cc[u][r];
        if (d < best[r]) { best[r] = d; bidx[r] = code; }
      }
    }
  }
  // cross-lane min over the 16 lanes of each half (xor<16 stays within half)
#pragma unroll
  for (int r = 0; r < 8; ++r) {
    float v = best[r]; int ix = bidx[r];
#pragma unroll
    for (int o = 1; o < 16; o <<= 1) {
      float ov = __shfl_xor(v, o, 32);
      int   oi = __shfl_xor(ix, o, 32);
      if (ov < v || (ov == v && oi < ix)) { v = ov; ix = oi; }
    }
    if (r0 == 0) { sval[wave * 16 + r + m0] = v; sidx[wave * 16 + r + m0] = ix; }
  }
  __syncthreads();
  if (t < 16) {
    float v = sval[t]; int ix = sidx[t];
    for (int w = 1; w < 4; ++w) {
      float ov = sval[w * 16 + t]; int oi = sidx[w * 16 + t];
      if (ov < v || (ov == v && oi < ix)) { v = ov; ix = oi; }
    }
    out_idx[rowbase + t] = ix;
    ws_idx[rowbase + t]  = ix;
  }
}

// ---------------- TDM-gather z_q + commit loss + GEMM3 ----------------
__global__ void k_post(const _Float16* __restrict__ ze, const _Float16* __restrict__ eh,
                       const int* __restrict__ idx, const _Float16* __restrict__ wpostT,
                       const float* __restrict__ bpost, float* __restrict__ outq,
                       float* __restrict__ commit) {
  __shared__ _Float16 As[16 * CDIM_];         // 32 KB z_q tile
  __shared__ float sred[128];
  int t = threadIdx.x, wave = t >> 5, lane = t & 31;
  long rowbase = (long)blockIdx.x * 16;
  if (t < 32) {
    // pack 16 row indices (uniform scalars) into descriptor groups 2/3
    int p[8];
#pragma unroll
    for (int j = 0; j < 8; ++j) {
      int i0 = idx[rowbase + 2 * j];
      int i1 = idx[rowbase + 2 * j + 1];
      p[j] = (i0 & 0xffff) | (i1 << 16);
    }
    i32x4 g2 = {p[0], p[1], p[2], p[3]};
    i32x4 g3 = {p[4], p[5], p[6], p[7]};
    tdm_gather16((unsigned int)(uintptr_t)As, eh, CDIM_, K_, CDIM_, g2, g3);
    __builtin_amdgcn_s_wait_tensorcnt(0);
  }
  __syncthreads();
  // commitment loss: sum (z_e - z_q)^2 ; z_q rows now sit in LDS
  float acc = 0.f;
#pragma unroll 1
  for (int i = 0; i < 16; ++i) {
    int chunk = t + i * 128;                  // 2048 chunks of 8 halves
    int row = chunk >> 7;
    int ko  = (chunk & 127) * 8;
    v8h z8 = *(const v8h*)(ze + (rowbase + row) * CDIM_ + ko);
    v8h e8 = *(const v8h*)(As + row * CDIM_ + ko);
#pragma unroll
    for (int u = 0; u < 8; ++u) {
      float d = (float)z8[u] - (float)e8[u];
      acc += d * d;
    }
  }
  sred[t] = acc;
  __syncthreads();
  for (int o = 64; o > 0; o >>= 1) { if (t < o) sred[t] += sred[t + o]; __syncthreads(); }
  if (t == 0) unsafeAtomicAdd(commit, sred[0] * (0.25f / ((float)ROWS_ * (float)CDIM_)));
  // GEMM3: z_q(16x1024) @ w_post(1024x256); 4 N-tiles per wave, shared A fragment
  int r0 = lane & 15, m0 = (lane & 16) ? 8 : 0, kboff = m0;
  const _Float16* arow = As + r0 * CDIM_;
  int col0 = wave * 4 * 16 + r0;
  const _Float16* brow = wpostT + (long)col0 * CDIM_;
  v8f cc[4] = {};
#pragma unroll 2
  for (int k0 = 0; k0 < CDIM_; k0 += 32) {
    v16h a = load_frag(arow, k0 + kboff);
#pragma unroll
    for (int u = 0; u < 4; ++u) {
      v16h b = load_frag(brow + u * 16 * CDIM_, k0 + kboff);
      cc[u] = WMMA(a, b, cc[u]);
    }
  }
#pragma unroll
  for (int u = 0; u < 4; ++u) {
    int col = col0 + u * 16;
    float bias = bpost[col];
#pragma unroll
    for (int r = 0; r < 8; ++r)
      outq[(rowbase + r + m0) * DIM_ + col] = cc[u][r] + bias;
  }
}

extern "C" void kernel_launch(void* const* d_in, const int* in_sizes, int n_in,
                              void* d_out, int out_size, void* d_ws, size_t ws_size,
                              hipStream_t stream) {
  const float* x      = (const float*)d_in[0];
  const float* ln_g   = (const float*)d_in[1];
  const float* ln_b   = (const float*)d_in[2];
  const float* w_pre  = (const float*)d_in[3];
  const float* b_pre  = (const float*)d_in[4];
  const float* embed  = (const float*)d_in[5];
  const float* w_post = (const float*)d_in[6];
  const float* b_post = (const float*)d_in[7];

  char* ws = (char*)d_ws;
  size_t off = 0;
  _Float16* h      = (_Float16*)(ws + off); off += (size_t)ROWS_ * DIM_ * 2;   // 32 MB
  _Float16* ze     = (_Float16*)(ws + off); off += (size_t)ROWS_ * CDIM_ * 2;  // 128 MB
  _Float16* eh     = (_Float16*)(ws + off); off += (size_t)K_ * CDIM_ * 2;     // 8 MB
  _Float16* wpreT  = (_Float16*)(ws + off); off += (size_t)CDIM_ * DIM_ * 2;
  _Float16* wpostT = (_Float16*)(ws + off); off += (size_t)DIM_ * CDIM_ * 2;
  float*    esq    = (float*)(ws + off);    off += (size_t)K_ * 4;
  int*      widx   = (int*)(ws + off);      off += (size_t)ROWS_ * 4;

  float* outq = (float*)d_out;                                  // 65536*256 f32
  int*   outi = (int*)((float*)d_out + (size_t)ROWS_ * DIM_);   // 65536 i32
  float* outc = (float*)d_out + (size_t)ROWS_ * DIM_ + ROWS_;   // 1 f32

  (void)hipMemsetAsync(outc, 0, sizeof(float), stream);
  k_prep_embed<<<K_, 256, 0, stream>>>(embed, eh, esq);
  k_prep_w<<<(DIM_ * CDIM_) / 256, 256, 0, stream>>>(w_pre, w_post, wpreT, wpostT);
  k_ln_gelu<<<ROWS_, 256, 0, stream>>>(x, ln_g, ln_b, h);
  k_gemm_pre<<<ROWS_ / 16, 256, 0, stream>>>(h, wpreT, b_pre, ze);
  k_dist_argmin<<<ROWS_ / 16, 128, 0, stream>>>(ze, eh, esq, outi, widx);
  k_post<<<ROWS_ / 16, 128, 0, stream>>>(ze, eh, widx, wpostT, b_post, outq, outc);
}
